// GroupedQueryAttention_20083267076165
// MI455X (gfx1250) — matrix operand order repaired and checked by reading, not compiled, not run on hardware
//
#include <hip/hip_runtime.h>
#include <stddef.h>

// ---------------------------------------------------------------------------
// GQA forward for MI455X (gfx1250, wave32, WMMA bf16 16x16x32, f32 accum)
//   D_MODEL=1024, N_HEADS=16, HEAD_DIM=64, KV_HEADS=4, B=2, S=2048
// Pipeline: cast X->bf16, transpose+cast W->bf16 (once, tiny), then
//   QKV GEMM -> flash attention -> out-proj GEMM, all WMMA bf16/f32-accum.
// ---------------------------------------------------------------------------

#define D_MODEL 1024
#define N_HEADS 16
#define HEAD_DIM 64
#define KV_HEADS 4
#define BATCH 2
#define SEQ 2048
#define MTOK (BATCH * SEQ)      // 4096

typedef __attribute__((ext_vector_type(16))) __bf16 v16bf;
typedef __attribute__((ext_vector_type(8)))  __bf16 v8bf;
typedef __attribute__((ext_vector_type(8)))  float  v8f;

// D = A(16x32 bf16) * B(32x16 bf16) + C(16x16 f32)
__device__ __forceinline__ v8f wmma_bf16(v16bf a, v16bf b, v8f c) {
  return __builtin_amdgcn_wmma_f32_16x16x32_bf16(
      false, a, false, b, (short)0, c, false, false);
}

// A-matrix 16x32 bf16 (ISA 7.12.2): lanes0-15 row=lane, K0-7 & K16-23;
// lanes16-31 row=lane-16, K8-15 & K24-31.  Two B128 loads per lane.
__device__ __forceinline__ v16bf load_a_bf16(const __bf16* src, int ld, int lane) {
  int row = lane & 15;
  int k0  = (lane & 16) ? 8 : 0;
  const __bf16* p = src + (size_t)row * ld + k0;
  v16bf a;
  ((v8bf*)&a)[0] = *(const v8bf*)(p);        // K k0..k0+7
  ((v8bf*)&a)[1] = *(const v8bf*)(p + 16);   // K k0+16..k0+23
  return a;
}

// B-matrix 32x16 bf16: lane&15 = N; lanes0-15 K0-7 & K16-23, lanes16-31 K8-15 & K24-31.
// Source element (k,n) at src[n*ldn + k] (k contiguous).  Two B128 loads.
__device__ __forceinline__ v16bf load_b_kcontig(const __bf16* src, int ldn, int lane) {
  int n  = lane & 15;
  int k0 = (lane & 16) ? 8 : 0;
  const __bf16* p = src + (size_t)n * ldn + k0;
  v16bf b;
  ((v8bf*)&b)[0] = *(const v8bf*)(p);
  ((v8bf*)&b)[1] = *(const v8bf*)(p + 16);
  return b;
}

// ---------------------------------------------------------------------------
// Kernel 0a: cast X (f32 row-major) -> bf16, vectorized 8/thread.
// ---------------------------------------------------------------------------
__global__ __launch_bounds__(256) void cast_x_kernel(
    const float* __restrict__ X, __bf16* __restrict__ Xb) {
  size_t i = ((size_t)blockIdx.x * blockDim.x + threadIdx.x) * 8;
  v8f v = *(const v8f*)(X + i);
  v8bf o;
#pragma unroll
  for (int j = 0; j < 8; ++j) o[j] = (__bf16)v[j];
  *(v8bf*)(Xb + i) = o;
}

// ---------------------------------------------------------------------------
// Kernel 0b: W[K x N] f32 -> Wt[N x K] bf16 (transpose + cast, one pass).
// ---------------------------------------------------------------------------
__global__ __launch_bounds__(256) void wt_kernel(
    const float* __restrict__ W, __bf16* __restrict__ Wt, int K, int N) {
  size_t idx = (size_t)blockIdx.x * blockDim.x + threadIdx.x;  // n*K + k, k fastest
  if (idx >= (size_t)K * N) return;
  int k = (int)(idx & (K - 1));      // K is a power of two (1024)
  int n = (int)(idx >> 10);
  Wt[idx] = (__bf16)W[(size_t)k * N + n];
}

// ---------------------------------------------------------------------------
// Kernel 1: QKV projection.  Xb[4096x1024]bf16 x Wt(bf16,[N][K]) -> Q/K/Vt bf16.
// One wave = 32(M) x 64(N) tile: 2 A-tiles share each B-tile.
// ---------------------------------------------------------------------------
__global__ __launch_bounds__(256) void qkv_proj_kernel(
    const __bf16* __restrict__ Xb,
    const __bf16* __restrict__ Wqt, const __bf16* __restrict__ Wkt,
    const __bf16* __restrict__ Wvt,
    const float* __restrict__ bq, const float* __restrict__ bk,
    const float* __restrict__ bv,
    __bf16* __restrict__ Qb, __bf16* __restrict__ Kb, __bf16* __restrict__ Vt) {
  int wave = blockIdx.x * (blockDim.x >> 5) + (threadIdx.x >> 5);
  int lane = threadIdx.x & 31;
  int mt = wave / 24;            // 0..127 (32-row tiles)
  int ns = wave % 24;            // 0..23  (64-col strips over 1536 fused cols)
  int m0 = mt * 32;
  int n0 = ns * 64;

  v8f acc[2][4];
#pragma unroll
  for (int u = 0; u < 2; ++u)
#pragma unroll
    for (int t = 0; t < 4; ++t) acc[u][t] = v8f{};

  for (int kk = 0; kk < D_MODEL; kk += 32) {
    v16bf a0 = load_a_bf16(Xb + (size_t)m0 * D_MODEL + kk, D_MODEL, lane);
    v16bf a1 = load_a_bf16(Xb + (size_t)(m0 + 16) * D_MODEL + kk, D_MODEL, lane);
#pragma unroll
    for (int t = 0; t < 4; ++t) {
      int nb = n0 + t * 16;                          // uniform per wave
      const __bf16* Wt;
      int nl;
      if (nb < 1024)      { Wt = Wqt; nl = nb; }
      else if (nb < 1280) { Wt = Wkt; nl = nb - 1024; }
      else                { Wt = Wvt; nl = nb - 1280; }
      v16bf b = load_b_kcontig(Wt + (size_t)nl * D_MODEL + kk, D_MODEL, lane);
      acc[0][t] = wmma_bf16(a0, b, acc[0][t]);
      acc[1][t] = wmma_bf16(a1, b, acc[1][t]);
    }
  }

  int n  = lane & 15;
  int rb = (lane & 16) ? 8 : 0;
#pragma unroll
  for (int u = 0; u < 2; ++u) {
#pragma unroll
    for (int t = 0; t < 4; ++t) {
      int nb = n0 + t * 16 + n;
      float bias = (nb < 1024) ? bq[nb] : (nb < 1280 ? bk[nb - 1024] : bv[nb - 1280]);
#pragma unroll
      for (int i = 0; i < 8; ++i) {
        int m = m0 + u * 16 + rb + i;
        int bidx = m >> 11;                 // / SEQ
        int s    = m & (SEQ - 1);
        __bf16 v = (__bf16)(acc[u][t][i] + bias);
        if (nb < 1024) {
          int h = nb >> 6, d = nb & 63;
          Qb[(((size_t)bidx * N_HEADS + h) * SEQ + s) * HEAD_DIM + d] = v;
        } else if (nb < 1280) {
          int c = nb - 1024, h = c >> 6, d = c & 63;
          Kb[(((size_t)bidx * KV_HEADS + h) * SEQ + s) * HEAD_DIM + d] = v;
        } else {
          int c = nb - 1280, h = c >> 6, d = c & 63;
          Vt[(((size_t)bidx * KV_HEADS + h) * HEAD_DIM + d) * SEQ + s] = v;  // V^T
        }
      }
    }
  }
}

// ---------------------------------------------------------------------------
// Kernel 2: causal flash attention (GQA). One wave per 16-query tile.
// ---------------------------------------------------------------------------
__global__ __launch_bounds__(256) void gqa_attn_kernel(
    const __bf16* __restrict__ Qb, const __bf16* __restrict__ Kb,
    const __bf16* __restrict__ Vt, __bf16* __restrict__ Ob) {
  __shared__ __bf16 Plds[8 * 16 * 32];     // per-wave 16x32 P staging
  int wid  = threadIdx.x >> 5;
  int lane = threadIdx.x & 31;
  int gw   = blockIdx.x * 8 + wid;         // 0..4095
  int qt   = gw & 127;
  int h    = (gw >> 7) & 15;
  int b    = gw >> 11;
  int kvh  = h >> 2;                       // jnp.repeat(K,4): head h -> kv head h/4

  const __bf16* Q  = Qb + (((size_t)b * N_HEADS + h) * SEQ + (size_t)qt * 16) * HEAD_DIM;
  const __bf16* Kh = Kb + ((size_t)b * KV_HEADS + kvh) * SEQ * HEAD_DIM;
  const __bf16* Vh = Vt + ((size_t)b * KV_HEADS + kvh) * HEAD_DIM * SEQ;
  __bf16* Pw = &Plds[wid * 512];

  v16bf aq0 = load_a_bf16(Q + 0,  HEAD_DIM, lane);
  v16bf aq1 = load_a_bf16(Q + 32, HEAD_DIM, lane);

  v8f o[4] = {v8f{}, v8f{}, v8f{}, v8f{}};
  float m_[8], l_[8];
#pragma unroll
  for (int i = 0; i < 8; ++i) { m_[i] = -1e30f; l_[i] = 0.0f; }

  int n  = lane & 15;
  int rb = (lane & 16) ? 8 : 0;
  int qend = qt * 16 + 15;

  for (int kb = 0; kb <= qend; kb += 32) {
    // scores: two 16x16 tiles over 32 keys (HD=64 -> 2 x K32 each)
    v8f s0 = v8f{}, s1 = v8f{};
    v16bf bk0 = load_b_kcontig(Kh + (size_t)kb * HEAD_DIM + 0,  HEAD_DIM, lane);
    v16bf bk1 = load_b_kcontig(Kh + (size_t)kb * HEAD_DIM + 32, HEAD_DIM, lane);
    s0 = wmma_bf16(aq0, bk0, s0);
    s0 = wmma_bf16(aq1, bk1, s0);
    v16bf bk2 = load_b_kcontig(Kh + (size_t)(kb + 16) * HEAD_DIM + 0,  HEAD_DIM, lane);
    v16bf bk3 = load_b_kcontig(Kh + (size_t)(kb + 16) * HEAD_DIM + 32, HEAD_DIM, lane);
    s1 = wmma_bf16(aq0, bk2, s1);
    s1 = wmma_bf16(aq1, bk3, s1);

    // online softmax (row reductions across 16-lane groups)
#pragma unroll
    for (int i = 0; i < 8; ++i) {
      int q = qt * 16 + rb + i;
      float v0 = s0[i] * 0.125f; if (kb + n      > q) v0 = -1e30f;
      float v1 = s1[i] * 0.125f; if (kb + 16 + n > q) v1 = -1e30f;
      float bm = fmaxf(v0, v1);
      bm = fmaxf(bm, __shfl_xor(bm, 1));
      bm = fmaxf(bm, __shfl_xor(bm, 2));
      bm = fmaxf(bm, __shfl_xor(bm, 4));
      bm = fmaxf(bm, __shfl_xor(bm, 8));
      float mn   = fmaxf(m_[i], bm);
      float corr = __expf(m_[i] - mn);
      float p0 = __expf(v0 - mn);
      float p1 = __expf(v1 - mn);
      float rs = p0 + p1;
      rs += __shfl_xor(rs, 1);
      rs += __shfl_xor(rs, 2);
      rs += __shfl_xor(rs, 4);
      rs += __shfl_xor(rs, 8);
      l_[i] = l_[i] * corr + rs;
      m_[i] = mn;
      o[0][i] *= corr; o[1][i] *= corr; o[2][i] *= corr; o[3][i] *= corr;
      Pw[(rb + i) * 32 + n]      = (__bf16)p0;   // C-layout -> row-major P
      Pw[(rb + i) * 32 + 16 + n] = (__bf16)p1;
    }
    asm volatile("s_wait_dscnt 0" ::: "memory");

    // O += P(16x32) x V(32x64); V^T layout makes B k-contiguous
    v16bf ap = load_a_bf16(Pw, 32, lane);
#pragma unroll
    for (int t = 0; t < 4; ++t) {
      v16bf bv = load_b_kcontig(Vh + (size_t)(t * 16) * SEQ + kb, SEQ, lane);
      o[t] = wmma_bf16(ap, bv, o[t]);
    }
    asm volatile("" ::: "memory");
  }

#pragma unroll
  for (int t = 0; t < 4; ++t) {
#pragma unroll
    for (int i = 0; i < 8; ++i) {
      float val = o[t][i] / l_[i];
      size_t tok = (size_t)b * SEQ + qt * 16 + rb + i;
      Ob[tok * D_MODEL + h * HEAD_DIM + t * 16 + n] = (__bf16)val;
    }
  }
}

// ---------------------------------------------------------------------------
// Kernel 3: output projection. Ob[4096x1024]bf16 x Wot(bf16 [N][K]) + bo -> f32
// One wave = 32(M) x 64(N).
// ---------------------------------------------------------------------------
__global__ __launch_bounds__(256) void out_proj_kernel(
    const __bf16* __restrict__ Ob, const __bf16* __restrict__ Wot,
    const float* __restrict__ bo, float* __restrict__ Out) {
  int wave = blockIdx.x * (blockDim.x >> 5) + (threadIdx.x >> 5);
  int lane = threadIdx.x & 31;
  int mt = wave >> 4;            // 0..127
  int ns = wave & 15;            // 0..15
  int m0 = mt * 32;
  int n0 = ns * 64;

  v8f acc[2][4];
#pragma unroll
  for (int u = 0; u < 2; ++u)
#pragma unroll
    for (int t = 0; t < 4; ++t) acc[u][t] = v8f{};

  for (int kk = 0; kk < D_MODEL; kk += 32) {
    v16bf a0 = load_a_bf16(Ob + (size_t)m0 * D_MODEL + kk, D_MODEL, lane);
    v16bf a1 = load_a_bf16(Ob + (size_t)(m0 + 16) * D_MODEL + kk, D_MODEL, lane);
#pragma unroll
    for (int t = 0; t < 4; ++t) {
      v16bf b = load_b_kcontig(Wot + (size_t)(n0 + t * 16) * D_MODEL + kk, D_MODEL, lane);
      acc[0][t] = wmma_bf16(a0, b, acc[0][t]);
      acc[1][t] = wmma_bf16(a1, b, acc[1][t]);
    }
  }

  int n  = lane & 15;
  int rb = (lane & 16) ? 8 : 0;
#pragma unroll
  for (int u = 0; u < 2; ++u)
#pragma unroll
    for (int t = 0; t < 4; ++t) {
      int nb = n0 + t * 16 + n;
      float bias = bo[nb];
#pragma unroll
      for (int i = 0; i < 8; ++i)
        Out[(size_t)(m0 + u * 16 + rb + i) * D_MODEL + nb] = acc[u][t][i] + bias;
    }
}

// ---------------------------------------------------------------------------
// Launch: inputs = {x, Wq, bq, Wk, bk, Wv, bv, Wo, bo, mask}
// ---------------------------------------------------------------------------
extern "C" void kernel_launch(void* const* d_in, const int* in_sizes, int n_in,
                              void* d_out, int out_size, void* d_ws, size_t ws_size,
                              hipStream_t stream) {
  const float* x  = (const float*)d_in[0];
  const float* Wq = (const float*)d_in[1];
  const float* bq = (const float*)d_in[2];
  const float* Wk = (const float*)d_in[3];
  const float* bk = (const float*)d_in[4];
  const float* Wv = (const float*)d_in[5];
  const float* bv = (const float*)d_in[6];
  const float* Wo = (const float*)d_in[7];
  const float* bo = (const float*)d_in[8];
  // d_in[9] = mask (causality computed analytically)

  // workspace (bf16 elements): Q | K | Vt | O | Xb | Wqt | Wkt | Wvt | Wot
  __bf16* Qb  = (__bf16*)d_ws;
  __bf16* Kb  = Qb  + (size_t)BATCH * N_HEADS * SEQ * HEAD_DIM;   // 4M
  __bf16* Vt  = Kb  + (size_t)BATCH * KV_HEADS * SEQ * HEAD_DIM;  // 1M
  __bf16* Ob  = Vt  + (size_t)BATCH * KV_HEADS * HEAD_DIM * SEQ;  // 1M
  __bf16* Xb  = Ob  + (size_t)MTOK * D_MODEL;                     // 4M
  __bf16* Wqt = Xb  + (size_t)MTOK * D_MODEL;                     // 4M
  __bf16* Wkt = Wqt + (size_t)D_MODEL * D_MODEL;                  // 1M
  __bf16* Wvt = Wkt + (size_t)D_MODEL * 256;                      // 256K
  __bf16* Wot = Wvt + (size_t)D_MODEL * 256;                      // 256K
  // total ~ 16.5M elems * 2B ~= 33 MB

  // Precast / pre-transpose (tiny, one-time per call)
  cast_x_kernel<<<(MTOK * D_MODEL / 8) / 256, 256, 0, stream>>>(x, Xb);
  wt_kernel<<<(D_MODEL * D_MODEL) / 256, 256, 0, stream>>>(Wq, Wqt, D_MODEL, D_MODEL);
  wt_kernel<<<(D_MODEL * 256) / 256, 256, 0, stream>>>(Wk, Wkt, D_MODEL, 256);
  wt_kernel<<<(D_MODEL * 256) / 256, 256, 0, stream>>>(Wv, Wvt, D_MODEL, 256);
  wt_kernel<<<(D_MODEL * D_MODEL) / 256, 256, 0, stream>>>(Wo, Wot, D_MODEL, D_MODEL);

  // QKV: 128 M-tiles(32) x 24 N-strips = 3072 waves -> 384 blocks of 8 waves
  qkv_proj_kernel<<<384, 256, 0, stream>>>(Xb, Wqt, Wkt, Wvt, bq, bk, bv, Qb, Kb, Vt);

  // Attention: 4096 query tiles -> 512 blocks of 8 waves
  gqa_attn_kernel<<<512, 256, 0, stream>>>(Qb, Kb, Vt, Ob);

  // Out projection: 128 x 16 = 2048 waves -> 256 blocks
  out_proj_kernel<<<256, 256, 0, stream>>>(Ob, Wot, bo, (float*)d_out);
}
